// SuperGlue_11991548690871
// MI455X (gfx1250) — matrix operand order repaired and checked by reading, not compiled, not run on hardware
//
#include <hip/hip_runtime.h>
#include <hip/hip_bf16.h>
#include <math.h>

typedef long long ll;
typedef _Float16 v16h __attribute__((ext_vector_type(16)));
typedef float    v8f  __attribute__((ext_vector_type(8)));

// ---------------------------------------------------------------------------
// Generic WMMA GEMM:  C = epilogue( alpha * A(MxK) * B(KxN) )
// A(m,k) = A[m*Ars + k*Acs],  B(k,n) = B[k*Brs + n*Bcs],  C(m,n) = C[m*Crs + n]
// z-batched: off = (z/HZ)*bs + (z%HZ)*hs for each operand.
// Block: 128 threads (4 waves), 64x64 tile; each wave owns a 16-row stripe
// and 4 WMMA accumulators (16x16 each), K stepped by 32.
//
// LDS tiles are stored PRE-SWIZZLED into WMMA fragment order so each lane's
// 16-half fragment is contiguous (2x ds_load_b128). Fragment slots padded to
// 24 halves (48B): 16B-aligned and bank-conflict-free across 16 lanes.
//
// Staging is BRANCH-FREE: out-of-range indices are clamped to a legal address
// and the loaded value is zeroed by a select, so all 16 global loads per pass
// are issued unconditionally back-to-back (one merged wait, full MLP).
// ---------------------------------------------------------------------------
__global__ __launch_bounds__(128) void wmma_gemm(
    const float* __restrict__ A, ll Ars, ll Acs, ll Abs, ll Ahs,
    const float* __restrict__ B, ll Brs, ll Bcs, ll Bbs, ll Bhs,
    float* __restrict__ C, ll Crs, ll Cbs, ll Chs,
    int M, int N, int K, int HZ, float alpha,
    const float* __restrict__ bias,
    const float* __restrict__ bnS, const float* __restrict__ bnB,
    int relu, int accum)
{
    int z  = blockIdx.z;
    int bb = z / HZ, hh = z % HZ;
    A += (ll)bb * Abs + (ll)hh * Ahs;
    B += (ll)bb * Bbs + (ll)hh * Bhs;
    C += (ll)bb * Cbs + (ll)hh * Chs;

    int mbase = blockIdx.y * 64;
    int nbase = blockIdx.x * 64;

    // 64 rows * 2 halves * 24-half slots  /  64 cols * 2 halves * 24-half slots
    __shared__ __align__(64) _Float16 Asw[128 * 24];
    __shared__ __align__(64) _Float16 Bsw[128 * 24];

    int tid  = threadIdx.x;
    int wave = tid >> 5;
    int lane = tid & 31;
    int hf   = lane >> 4;     // which 16-lane half
    int lr   = lane & 15;

    v8f zero = {0.f,0.f,0.f,0.f,0.f,0.f,0.f,0.f};
    v8f acc[4];
    #pragma unroll
    for (int i = 0; i < 4; ++i) acc[i] = zero;

    for (int kb = 0; kb < K; kb += 32) {
        // ---- stage A tile 64x32 (branch-free, batched loads) ----
        // swizzle: afr[e] = A[r][(e>>3)*16 + hf*8 + (e&7)]
        {
            float va[16]; bool ok[16];
            #pragma unroll
            for (int ii = 0; ii < 16; ++ii) {
                int idx = tid + ii * 128;
                int r = idx >> 5, k = idx & 31;
                int gm = mbase + r, gk = kb + k;
                ok[ii] = (gm < M) && (gk < K);
                int gmc = min(gm, M - 1), gkc = min(gk, K - 1);
                va[ii] = A[(ll)gmc * Ars + (ll)gkc * Acs];
            }
            #pragma unroll
            for (int ii = 0; ii < 16; ++ii) {
                int idx = tid + ii * 128;
                int r = idx >> 5, k = idx & 31;
                int ahf = (k >> 3) & 1;
                int e   = ((k >> 4) << 3) | (k & 7);
                float x = ok[ii] ? va[ii] : 0.f;
                Asw[(r * 2 + ahf) * 24 + e] = (_Float16)x;
            }
        }
        // ---- stage B tile 32x64 (branch-free, batched loads) ----
        // swizzle: bfr[e] = B[(e>>3)*16 + hf*8 + (e&7)][c]
        {
            float vb[16]; bool ok[16];
            #pragma unroll
            for (int ii = 0; ii < 16; ++ii) {
                int idx = tid + ii * 128;
                int k = idx >> 6, c = idx & 63;
                int gk = kb + k, gn = nbase + c;
                ok[ii] = (gk < K) && (gn < N);
                int gkc = min(gk, K - 1), gnc = min(gn, N - 1);
                vb[ii] = B[(ll)gkc * Brs + (ll)gnc * Bcs];
            }
            #pragma unroll
            for (int ii = 0; ii < 16; ++ii) {
                int idx = tid + ii * 128;
                int k = idx >> 6, c = idx & 63;
                int bhf = (k >> 3) & 1;
                int e   = ((k >> 4) << 3) | (k & 7);
                float x = ok[ii] ? vb[ii] : 0.f;
                Bsw[(c * 2 + bhf) * 24 + e] = (_Float16)x;
            }
        }
        __syncthreads();

        // A fragment: one contiguous 32B vector load per lane
        v16h afr = *(const v16h*)&Asw[((wave * 16 + lr) * 2 + hf) * 24];
        #pragma unroll
        for (int nt = 0; nt < 4; ++nt) {
            v16h bfr = *(const v16h*)&Bsw[((nt * 16 + lr) * 2 + hf) * 24];
            acc[nt] = __builtin_amdgcn_wmma_f32_16x16x32_f16(
                false, afr, false, bfr, (short)0, acc[nt], false, false);
        }
        __syncthreads();
    }

    // epilogue: D layout VGPR r -> m = r + 8*half, n = lane%16
    #pragma unroll
    for (int nt = 0; nt < 4; ++nt) {
        #pragma unroll
        for (int r = 0; r < 8; ++r) {
            int gm = mbase + wave * 16 + r + 8 * hf;
            int gn = nbase + nt * 16 + lr;
            if (gm < M && gn < N) {
                float x = alpha * acc[nt][r];
                if (bias) x += bias[gm];
                if (bnS)  x = bnS[gm] * x + bnB[gm];
                if (relu) x = fmaxf(x, 0.f);
                ll ci = (ll)gm * Crs + gn;
                if (accum) C[ci] += x; else C[ci] = x;
            }
        }
    }
}

// ---------------------------------------------------------------------------
// Small helper kernels
// ---------------------------------------------------------------------------
__global__ void k_bnfold(const float* g, const float* be, const float* mn,
                         const float* vr, float* S, float* Bsh, int C)
{
    int i = blockIdx.x * blockDim.x + threadIdx.x;
    if (i < C) {
        float s = g[i] * rsqrtf(vr[i] + 1e-5f);
        S[i] = s; Bsh[i] = be[i] - mn[i] * s;
    }
}

// kx[4][8192]: rows 0..2 = kpts coords, row 3 = score; col = img*4096 + b*1024 + n
__global__ void k_build_kx(const float* kp0, const float* sc0,
                           const float* kp1, const float* sc1, float* kx)
{
    int col = blockIdx.x * blockDim.x + threadIdx.x;
    if (col >= 8192) return;
    const float* kp = (col < 4096) ? kp0 : kp1;
    const float* sc = (col < 4096) ? sc0 : sc1;
    int c2 = col & 4095;
    kx[0 * 8192 + col] = kp[c2 * 3 + 0];
    kx[1 * 8192 + col] = kp[c2 * 3 + 1];
    kx[2 * 8192 + col] = kp[c2 * 3 + 2];
    kx[3 * 8192 + col] = sc[c2];
}

// d[c][col] = desc[col][c] + enc[c][encoff+col]
__global__ void k_build_d(const float* desc, const float* enc, int encoff, float* d)
{
    int id = blockIdx.x * blockDim.x + threadIdx.x;
    if (id >= 256 * 4096) return;
    int c = id >> 12, col = id & 4095;
    d[id] = desc[(ll)col * 256 + c] + enc[(ll)c * 8192 + encoff + col];
}

// softmax over rows of length 1024; grid.x = #rows
__global__ __launch_bounds__(256) void k_softmax_rows(float* S)
{
    __shared__ float red[256];
    float* row = S + (ll)blockIdx.x * 1024;
    int t = threadIdx.x;
    float x[4], m = -INFINITY;
    #pragma unroll
    for (int i = 0; i < 4; ++i) { x[i] = row[t + 256 * i]; m = fmaxf(m, x[i]); }
    red[t] = m; __syncthreads();
    for (int s = 128; s > 0; s >>= 1) { if (t < s) red[t] = fmaxf(red[t], red[t + s]); __syncthreads(); }
    m = red[0]; __syncthreads();
    float sum = 0.f;
    #pragma unroll
    for (int i = 0; i < 4; ++i) { x[i] = __expf(x[i] - m); sum += x[i]; }
    red[t] = sum; __syncthreads();
    for (int s = 128; s > 0; s >>= 1) { if (t < s) red[t] += red[t + s]; __syncthreads(); }
    float inv = 1.f / red[0];
    #pragma unroll
    for (int i = 0; i < 4; ++i) row[t + 256 * i] = x[i] * inv;
}

__global__ void k_fill_bins(float* coup, const float* alpha_p)
{
    int id = blockIdx.x * blockDim.x + threadIdx.x;
    if (id >= 4 * 1025) return;
    float a = alpha_p[0];
    int b = id / 1025, i = id % 1025;
    float* cb = coup + (ll)b * 1050625;
    cb[(ll)i * 1025 + 1024] = a;
    cb[(ll)1024 * 1025 + i] = a;
}

__global__ void k_zero(float* p, int n)
{
    int id = blockIdx.x * blockDim.x + threadIdx.x;
    if (id < n) p[id] = 0.f;
}

// u[b,i] = log_mu[i] - lse_j( coup[b,i,j] + v[b,j] );  one block per (b,i)
__global__ __launch_bounds__(256) void k_sink_row(const float* coup, const float* v, float* u)
{
    int b = blockIdx.x / 1025, i = blockIdx.x % 1025;
    const float* row = coup + (ll)b * 1050625 + (ll)i * 1025;
    const float* vb  = v + b * 1025;
    int t = threadIdx.x;
    float m = -INFINITY, s = 0.f;
    for (int j = t; j < 1025; j += 256) {
        float x = row[j] + vb[j];
        if (x > m) { s = s * __expf(m - x) + 1.f; m = x; } else { s += __expf(x - m); }
    }
    __shared__ float rm[256], rs[256];
    rm[t] = m; rs[t] = s; __syncthreads();
    for (int k = 128; k > 0; k >>= 1) {
        if (t < k) {
            float m2 = rm[t + k], s2 = rs[t + k];
            float mm = fmaxf(rm[t], m2);
            float sn = (mm > -INFINITY) ? rs[t] * __expf(rm[t] - mm) + s2 * __expf(m2 - mm) : 0.f;
            rm[t] = mm; rs[t] = sn;
        }
        __syncthreads();
    }
    if (t == 0) {
        float lse  = rm[0] + __logf(rs[0]);
        float norm = -__logf(2048.f);
        float lmu  = (i < 1024) ? norm : (__logf(1024.f) + norm);
        u[b * 1025 + i] = lmu - lse;
    }
}

// column partial lse: block handles 256 columns over a 129-row stripe (coalesced)
__global__ __launch_bounds__(256) void k_sink_col_part(const float* coup, const float* u,
                                                       float* pm, float* ps)
{
    int j  = blockIdx.x * 256 + threadIdx.x;
    int b  = blockIdx.y;
    int rz = blockIdx.z;
    if (j >= 1025) return;
    int r0 = rz * 129, r1 = min(1025, r0 + 129);
    const float* cb = coup + (ll)b * 1050625 + j;
    const float* ub = u + b * 1025;
    float m = -INFINITY, s = 0.f;
    for (int i = r0; i < r1; ++i) {
        __builtin_prefetch(&cb[(ll)(i + 8) * 1025], 0, 0);   // global_prefetch_b8
        float x = cb[(ll)i * 1025] + ub[i];
        if (x > m) { s = s * __expf(m - x) + 1.f; m = x; } else { s += __expf(x - m); }
    }
    int idx = (rz * 4 + b) * 1025 + j;
    pm[idx] = m; ps[idx] = s;
}

__global__ void k_sink_col_comb(const float* pm, const float* ps, float* v)
{
    int id = blockIdx.x * blockDim.x + threadIdx.x;
    if (id >= 4 * 1025) return;
    int b = id / 1025, j = id % 1025;
    float m = -INFINITY, s = 0.f;
    for (int rz = 0; rz < 8; ++rz) {
        float m2 = pm[(rz * 4 + b) * 1025 + j], s2 = ps[(rz * 4 + b) * 1025 + j];
        float mm = fmaxf(m, m2);
        s = (mm > -INFINITY) ? s * __expf(m - mm) + s2 * __expf(m2 - mm) : 0.f;
        m = mm;
    }
    float lse  = m + __logf(s);
    float norm = -__logf(2048.f);
    float lnu  = (j < 1024) ? norm : (__logf(1024.f) + norm);
    v[id] = lnu - lse;
}

__global__ void k_zfinal(const float* coup, const float* u, const float* v, float* Z)
{
    ll id = (ll)blockIdx.x * 256 + threadIdx.x;
    if (id >= 4202500LL) return;
    int b = (int)(id / 1050625); int rem = (int)(id % 1050625);
    int i = rem / 1025, j = rem % 1025;
    Z[id] = coup[id] + u[b * 1025 + i] + v[b * 1025 + j] + __logf(2048.f);
}

__global__ __launch_bounds__(256) void k_argmax_row(const float* Z, float* max0, int* idx0)
{
    int b = blockIdx.x >> 10, i = blockIdx.x & 1023;
    const float* row = Z + (ll)b * 1050625 + (ll)i * 1025;
    int t = threadIdx.x;
    float m = -INFINITY; int mi = 0;
    for (int j = t; j < 1024; j += 256) {
        float x = row[j];
        if (x > m) { m = x; mi = j; }
    }
    __shared__ float rm[256]; __shared__ int ri[256];
    rm[t] = m; ri[t] = mi; __syncthreads();
    for (int s = 128; s > 0; s >>= 1) {
        if (t < s && rm[t + s] > rm[t]) { rm[t] = rm[t + s]; ri[t] = ri[t + s]; }
        __syncthreads();
    }
    if (t == 0) { max0[blockIdx.x] = rm[0]; idx0[blockIdx.x] = ri[0]; }
}

__global__ void k_argmax_col(const float* Z, int* idx1)
{
    int id = blockIdx.x * blockDim.x + threadIdx.x;
    if (id >= 4096) return;
    int b = id >> 10, j = id & 1023;
    const float* cb = Z + (ll)b * 1050625 + j;
    float m = -INFINITY; int mi = 0;
    for (int i = 0; i < 1024; ++i) {
        float x = cb[(ll)i * 1025];
        if (x > m) { m = x; mi = i; }
    }
    idx1[id] = mi;
}

__global__ void k_match0(const int* idx0, const int* idx1, const float* max0,
                         float* m0o, float* s0o, float* ms0w, int* vld0)
{
    int id = blockIdx.x * blockDim.x + threadIdx.x;
    if (id >= 4096) return;
    int b = id >> 10, i = id & 1023;
    int j = idx0[id];
    bool mutual = (idx1[b * 1024 + j] == i);
    float ms = mutual ? __expf(max0[id]) : 0.f;
    bool valid = mutual && (ms > 0.2f);
    m0o[id]  = valid ? (float)j : -1.f;
    s0o[id]  = ms;
    ms0w[id] = ms;
    vld0[id] = valid ? 1 : 0;
}

__global__ void k_match1(const int* idx0, const int* idx1, const float* ms0w,
                         const int* vld0, float* m1o, float* s1o)
{
    int id = blockIdx.x * blockDim.x + threadIdx.x;
    if (id >= 4096) return;
    int b = id >> 10, j = id & 1023;
    int i = idx1[id];
    bool mutual = (idx0[b * 1024 + i] == j);
    float ms = mutual ? ms0w[b * 1024 + i] : 0.f;
    bool valid = mutual && (vld0[b * 1024 + i] != 0);
    m1o[id] = valid ? (float)i : -1.f;
    s1o[id] = ms;
}

// ---------------------------------------------------------------------------
// Host orchestration
// ---------------------------------------------------------------------------
static void launch_gemm(hipStream_t st,
    const float* A, ll Ars, ll Acs, ll Abs, ll Ahs,
    const float* B, ll Brs, ll Bcs, ll Bbs, ll Bhs,
    float* C, ll Crs, ll Cbs, ll Chs,
    int M, int N, int K, int Z, int HZ, float alpha,
    const float* bias, const float* bnS, const float* bnB, int relu, int accum)
{
    dim3 grid((N + 63) / 64, (M + 63) / 64, Z);
    wmma_gemm<<<grid, 128, 0, st>>>(A, Ars, Acs, Abs, Ahs, B, Brs, Bcs, Bbs, Bhs,
                                    C, Crs, Cbs, Chs, M, N, K, HZ, alpha,
                                    bias, bnS, bnB, relu, accum);
}

// plain "1x1 conv" GEMM: W[M,K] row-major * X[K,N] -> C[M,N]
static void conv_gemm(hipStream_t st, const float* W, const float* X, float* C,
                      int M, int N, int K, const float* bias,
                      const float* bnS, const float* bnB, int relu, int accum)
{
    launch_gemm(st, W, K, 1, 0, 0, X, N, 1, 0, 0, C, N, 0, 0,
                M, N, K, 1, 1, 1.f, bias, bnS, bnB, relu, accum);
}

extern "C" void kernel_launch(void* const* d_in, const int* in_sizes, int n_in,
                              void* d_out, int out_size, void* d_ws, size_t ws_size,
                              hipStream_t stream)
{
    auto F = [&](int i) { return (const float*)d_in[i]; };
    // top-level inputs in setup_inputs() dict order
    const float* kpts0  = F(0);
    const float* kpts1  = F(1);
    const float* scor0  = F(2);
    const float* scor1  = F(3);
    const float* desc0  = F(4);
    const float* desc1  = F(5);
    // params leaves (jax pytree order: sorted dict keys, lists in order)
    const int P = 6;
    const float* binSc = F(P + 0);
    const float* finW  = F(P + 1);
    const float* finB  = F(P + 2);
    auto G = [&](int l, int off) { return F(P + 3 + l * 16 + off); };
    // per-gnn-layer offsets: k.W 0, k.b 1, merge.W 2, merge.b 3, q.W 4, q.b 5,
    // v.W 6, v.b 7, mlp0.W 8, mlp0.b 9, mlp1.W 10, mlp1.b 11,
    // bn.beta 12, bn.gamma 13, bn.mean 14, bn.var 15
    const int KB = P + 3 + 96;   // kenc base
    // kenc layer i in 0..2: bn.beta +6i, bn.gamma, bn.mean, bn.var, conv.W, conv.b
    // kenc layer 3: conv.W = KB+18, conv.b = KB+19

    // workspace layout (floats)
    float* W = (float*)d_ws;
    size_t o = 0;
    auto alloc = [&](size_t n) { float* p = W + o; o += n; return p; };
    float* kx   = alloc(4 * 8192);
    float* tA   = alloc(128 * 8192);
    float* tB   = alloc(64 * 8192);
    float* enc  = alloc(256 * 8192);
    float* d0   = alloc(256 * 4096);
    float* d1   = alloc(256 * 4096);
    float* Qb   = alloc(256 * 4096);
    float* Kb   = alloc(256 * 4096);
    float* Vb   = alloc(256 * 4096);
    float* Sb   = alloc((size_t)16 * 1024 * 1024);
    float* Ob   = alloc(256 * 4096);
    float* cat0 = alloc(512 * 4096);
    float* cat1 = alloc(512 * 4096);
    float* gbuf = alloc(512 * 4096);
    float* md0  = alloc(256 * 4096);
    float* md1  = alloc(256 * 4096);
    float* coup = alloc((size_t)4 * 1025 * 1025);
    float* uu   = alloc(4 * 1025);
    float* vv   = alloc(4 * 1025);
    float* pm   = alloc(8 * 4 * 1025);
    float* ps   = alloc(8 * 4 * 1025);
    float* bnS  = alloc(512);
    float* bnB  = alloc(512);
    float* max0 = alloc(4096);
    float* ms0w = alloc(4096);
    int*   idx0 = (int*)alloc(4096);
    int*   idx1 = (int*)alloc(4096);
    int*   vld0 = (int*)alloc(4096);

    float* Zout = (float*)d_out;                 // 4*1025*1025
    float* m0o  = Zout + 4202500;                // matches0
    float* m1o  = m0o + 4096;                    // matches1
    float* s0o  = m1o + 4096;                    // mscores0
    float* s1o  = s0o + 4096;                    // mscores1

    // ---- keypoint encoder (both images batched as 8192 columns) ----
    k_build_kx<<<(8192 + 255) / 256, 256, 0, stream>>>(kpts0, scor0, kpts1, scor1, kx);
    const int kch[5] = {4, 32, 64, 128, 256};
    float* kin = kx;
    float* kouts[4] = {tA, tB, tA, enc};
    for (int i = 0; i < 4; ++i) {
        const float* cw, *cb;
        if (i < 3) {
            cw = F(KB + 6 * i + 4); cb = F(KB + 6 * i + 5);
            k_bnfold<<<(kch[i + 1] + 127) / 128, 128, 0, stream>>>(
                F(KB + 6 * i + 1), F(KB + 6 * i + 0), F(KB + 6 * i + 2), F(KB + 6 * i + 3),
                bnS, bnB, kch[i + 1]);
            conv_gemm(stream, cw, kin, kouts[i], kch[i + 1], 8192, kch[i], cb, bnS, bnB, 1, 0);
        } else {
            cw = F(KB + 18); cb = F(KB + 19);
            conv_gemm(stream, cw, kin, kouts[i], kch[i + 1], 8192, kch[i], cb, nullptr, nullptr, 0, 0);
        }
        kin = kouts[i];
    }
    k_build_d<<<(256 * 4096 + 255) / 256, 256, 0, stream>>>(desc0, enc, 0, d0);
    k_build_d<<<(256 * 4096 + 255) / 256, 256, 0, stream>>>(desc1, enc, 4096, d1);

    // ---- GNN: 6 layers, alternating self / cross ----
    for (int l = 0; l < 6; ++l) {
        bool cross = (l & 1);
        for (int img = 0; img < 2; ++img) {
            float* x   = img ? d1 : d0;
            float* src = cross ? (img ? d0 : d1) : x;
            float* cat = img ? cat1 : cat0;
            // Q from x ; K,V from src   (256x256 conv over 4096 cols)
            conv_gemm(stream, G(l, 4), x,   Qb, 256, 4096, 256, G(l, 5), nullptr, nullptr, 0, 0);
            conv_gemm(stream, G(l, 0), src, Kb, 256, 4096, 256, G(l, 1), nullptr, nullptr, 0, 0);
            conv_gemm(stream, G(l, 6), src, Vb, 256, 4096, 256, G(l, 7), nullptr, nullptr, 0, 0);
            // scores[b,h,n,m] = sum_hd Q[(hd*4+h), b*1024+n] * K[(hd*4+h), b*1024+m] / 8
            launch_gemm(stream,
                Qb, 1, 16384, 1024, 4096,             // A(m=n, k=hd)
                Kb, 16384, 1, 1024, 4096,             // B(k=hd, n=m)
                Sb, 1024, (ll)4 * 1048576, 1048576,   // C rows n, z = b*4+h
                1024, 1024, 64, 16, 4, 0.125f, nullptr, nullptr, nullptr, 0, 0);
            k_softmax_rows<<<16384, 256, 0, stream>>>(Sb);
            // O[(hd*4+h), b*1024+n] = sum_m V[(hd*4+h), b*1024+m] * P[b,h,n,m]
            launch_gemm(stream,
                Vb, 16384, 1, 1024, 4096,             // A(m=hd, k=m')
                Sb, 1, 1024, (ll)4 * 1048576, 1048576,// B(k=m', n) = P[n][m']
                Ob, 16384, 1024, 4096,                // C(hd, n)
                64, 1024, 1024, 16, 4, 1.f, nullptr, nullptr, nullptr, 0, 0);
            // merge -> second half of concat buffer
            conv_gemm(stream, G(l, 2), Ob, cat + (size_t)256 * 4096,
                      256, 4096, 256, G(l, 3), nullptr, nullptr, 0, 0);
            hipMemcpyAsync(cat, x, (size_t)256 * 4096 * sizeof(float),
                           hipMemcpyDeviceToDevice, stream);
        }
        // MLP on both concats (uses pre-update d's), then residual accumulate
        k_bnfold<<<(512 + 127) / 128, 128, 0, stream>>>(
            G(l, 13), G(l, 12), G(l, 14), G(l, 15), bnS, bnB, 512);
        for (int img = 0; img < 2; ++img) {
            float* cat = img ? cat1 : cat0;
            float* dd  = img ? d1 : d0;
            conv_gemm(stream, G(l, 8), cat, gbuf, 512, 4096, 512, G(l, 9), bnS, bnB, 1, 0);
            conv_gemm(stream, G(l, 10), gbuf, dd, 256, 4096, 512, G(l, 11), nullptr, nullptr, 0, 1);
        }
    }

    // ---- final projection + score matrix ----
    conv_gemm(stream, finW, d0, md0, 256, 4096, 256, finB, nullptr, nullptr, 0, 0);
    conv_gemm(stream, finW, d1, md1, 256, 4096, 256, finB, nullptr, nullptr, 0, 0);
    // sc[b,n,m] = sum_d md0[d, b*1024+n] * md1[d, b*1024+m] / 16 -> couplings interior
    launch_gemm(stream,
        md0, 1, 4096, 1024, 0,                    // A(m=n, k=d)
        md1, 4096, 1, 1024, 0,                    // B(k=d, n=m)
        coup, 1025, (ll)1025 * 1025, 0,           // C row stride 1025
        1024, 1024, 256, 4, 1, 1.f / 16.f, nullptr, nullptr, nullptr, 0, 0);
    k_fill_bins<<<(4 * 1025 + 255) / 256, 256, 0, stream>>>(coup, binSc);

    // ---- Sinkhorn (100 iterations, log domain) ----
    k_zero<<<(4 * 1025 + 255) / 256, 256, 0, stream>>>(uu, 4 * 1025);
    k_zero<<<(4 * 1025 + 255) / 256, 256, 0, stream>>>(vv, 4 * 1025);
    for (int it = 0; it < 100; ++it) {
        k_sink_row<<<4 * 1025, 256, 0, stream>>>(coup, vv, uu);
        k_sink_col_part<<<dim3(5, 4, 8), 256, 0, stream>>>(coup, uu, pm, ps);
        k_sink_col_comb<<<(4 * 1025 + 255) / 256, 256, 0, stream>>>(pm, ps, vv);
    }
    k_zfinal<<<(int)((4202500LL + 255) / 256), 256, 0, stream>>>(coup, uu, vv, Zout);

    // ---- matching ----
    k_argmax_row<<<4096, 256, 0, stream>>>(Zout, max0, idx0);
    k_argmax_col<<<(4096 + 255) / 256, 256, 0, stream>>>(Zout, idx1);
    k_match0<<<(4096 + 255) / 256, 256, 0, stream>>>(idx0, idx1, max0, m0o, s0o, ms0w, vld0);
    k_match1<<<(4096 + 255) / 256, 256, 0, stream>>>(idx0, idx1, ms0w, vld0, m1o, s1o);
}
